// AttentionLayer_5669356830757
// MI455X (gfx1250) — compile-verified, matrix operand order checked by reading, not run on hardware
//
#include <hip/hip_runtime.h>
#include <hip/hip_bf16.h>
#include <stdint.h>

// ---------------------------------------------------------------------------
// MI455X (gfx1250) attention layer: all matmuls via V_WMMA_F32_16X16X32_BF16.
// f32 inputs -> bf16 operands (f32 accumulate), flash-style online softmax.
// Round 2: 32x64 per-wave GEMM tiles (B reuse x2), 64-key attention rounds,
// QKV fused into one launch, softmax scale folded into Q projection.
// ---------------------------------------------------------------------------

typedef __bf16 bf16;
typedef __attribute__((ext_vector_type(8)))  bf16  v8bf;
typedef __attribute__((ext_vector_type(16))) bf16  v16bf;
typedef __attribute__((ext_vector_type(8)))  float v8f;

constexpr int Bq = 2, Sq = 2048, Dm = 1024, Hh = 16, HDd = 64;
constexpr int Mtot = Bq * Sq;   // 4096 rows

#define DEV static __device__ __forceinline__

DEV bf16 f2bf(float f) {
  unsigned u = __builtin_bit_cast(unsigned, f);
  unsigned r = u + 0x7FFFu + ((u >> 16) & 1u);   // round-to-nearest-even
  return __builtin_bit_cast(bf16, (unsigned short)(r >> 16));
}

DEV v16bf cat16(v8bf lo, v8bf hi) {
  return __builtin_shufflevector(lo, hi, 0,1,2,3,4,5,6,7,8,9,10,11,12,13,14,15);
}

DEV v16bf ldab(const bf16* p) {          // one WMMA A/B operand: 2x16B loads
  return cat16(*(const v8bf*)p, *(const v8bf*)(p + 16));
}

DEV v8f wmma_bf16(v16bf a, v16bf b, v8f c) {
  // (neg_a, A, neg_b, B, c_mod, C, reuse_a, reuse_b)
  return __builtin_amdgcn_wmma_f32_16x16x32_bf16(false, a, false, b, (short)0, c,
                                                 false, false);
}

// ---------------------------------------------------------------------------
// f32 -> bf16 bulk convert
// ---------------------------------------------------------------------------
__global__ void __launch_bounds__(256) cvt_kernel(const float* __restrict__ in,
                                                  bf16* __restrict__ out, int n) {
  int i = blockIdx.x * blockDim.x + threadIdx.x;
  int stride = gridDim.x * blockDim.x;
  for (; i < n; i += stride) out[i] = f2bf(in[i]);
}

// ---------------------------------------------------------------------------
// 1024x1024 f32 -> bf16 transpose (weights become N-major so the WMMA B
// operand reads contiguous K runs). LDS-tiled for coalescing.
// ---------------------------------------------------------------------------
__global__ void __launch_bounds__(256) transpose_cvt_kernel(
    const float* __restrict__ src, bf16* __restrict__ dst) {
  __shared__ float tile[32][33];
  const int x = blockIdx.x * 32 + threadIdx.x;
  const int y = blockIdx.y * 32 + threadIdx.y;
#pragma unroll
  for (int i = 0; i < 32; i += 8)
    tile[threadIdx.y + i][threadIdx.x] = src[(size_t)(y + i) * Dm + x];
  __syncthreads();
  const int tx = blockIdx.y * 32 + threadIdx.x;   // transposed col (= old row)
  const int ty = blockIdx.x * 32 + threadIdx.y;   // transposed row (= old col)
#pragma unroll
  for (int i = 0; i < 32; i += 8)
    dst[(size_t)(ty + i) * Dm + tx] = f2bf(tile[threadIdx.x][threadIdx.y + i]);
}

// ---------------------------------------------------------------------------
// Fused QKV projection GEMM: [4096 x 1024] x [1024 x 1024] (Wt N-major).
// blockIdx.z in {0,1,2} selects Q / K / V.  Per wave: 32(M) x 64(N) tile,
// K step 32 -> 8 WMMAs/step, each B operand reused by 2 WMMAs.
// Q epilogue folds the 1/sqrt(hd)=0.125 softmax scale (exact in bf16).
// Q,K written [bh][s][hd]; V written transposed [bh][hd][s].
// ---------------------------------------------------------------------------
__global__ void __launch_bounds__(128) qkv_gemm_kernel(
    const bf16* __restrict__ X,
    const bf16* __restrict__ WqT, const bf16* __restrict__ WkT,
    const bf16* __restrict__ WvT,
    const float* __restrict__ bqv, const float* __restrict__ bkv,
    const float* __restrict__ bvv,
    bf16* __restrict__ Qo, bf16* __restrict__ Ko, bf16* __restrict__ Vo) {
  const int z = blockIdx.z;
  const bf16*  Wt   = (z == 0) ? WqT : (z == 1) ? WkT : WvT;
  const float* bias = (z == 0) ? bqv : (z == 1) ? bkv : bvv;
  bf16*        out  = (z == 0) ? Qo  : (z == 1) ? Ko  : Vo;
  const float  postscale = (z == 0) ? 0.125f : 1.0f;

  const int lane  = threadIdx.x & 31;
  const int wave  = threadIdx.x >> 5;
  const int lhalf = lane & 15;
  const int koff  = (lane < 16) ? 0 : 8;       // 16-bit A/B lane striping
  const int m0 = blockIdx.x * 128 + wave * 32;
  const int n0 = blockIdx.y * 64;

  v8f acc[2][4];
#pragma unroll
  for (int r = 0; r < 2; r++)
#pragma unroll
    for (int j = 0; j < 4; j++)
#pragma unroll
      for (int e = 0; e < 8; e++) acc[r][j][e] = 0.f;

  const bf16* arow0 = X + (size_t)(m0 + lhalf) * Dm;
  const bf16* arow1 = X + (size_t)(m0 + 16 + lhalf) * Dm;
  for (int k0 = 0; k0 < Dm; k0 += 32) {
    v16bf a0 = ldab(arow0 + k0 + koff);
    v16bf a1 = ldab(arow1 + k0 + koff);
#pragma unroll
    for (int j = 0; j < 4; j++) {
      const bf16* brow = Wt + (size_t)(n0 + j * 16 + lhalf) * Dm;
      v16bf b = ldab(brow + k0 + koff);
      acc[0][j] = wmma_bf16(a0, b, acc[0][j]);
      acc[1][j] = wmma_bf16(a1, b, acc[1][j]);
    }
  }

  // C/D layout: VGPR e = row e (lanes 0-15) / row e+8 (lanes 16-31)
#pragma unroll
  for (int r = 0; r < 2; r++) {
    const int rbase = m0 + r * 16 + ((lane < 16) ? 0 : 8);
#pragma unroll
    for (int j = 0; j < 4; j++) {
      const int col = n0 + j * 16 + lhalf;     // n in [0,1024)
      const int h = col >> 6, hd = col & 63;
      const float bb = bias[col];
#pragma unroll
      for (int e = 0; e < 8; e++) {
        const int m = rbase + e;               // b*S + s
        const int bidx = m >> 11, s = m & 2047;
        const float v = (acc[r][j][e] + bb) * postscale;
        size_t idx;
        if (z == 2)   // V transposed: [bh][hd][s]
          idx = ((size_t)(bidx * Hh + h) * HDd + hd) * Sq + s;
        else          // Q / K: [bh][s][hd]
          idx = ((size_t)(bidx * Hh + h) * Sq + s) * HDd + hd;
        out[idx] = f2bf(v);
      }
    }
  }
}

// ---------------------------------------------------------------------------
// Flash attention per (b,h): wave owns 16 q-rows; 64 keys per round.
//   scores: 8 WMMAs (4 key-chunks x 2 hd-chunks), online softmax (Q is
//   pre-scaled), P -> LDS (D-layout store, A-layout reload), PV: 8 WMMAs.
// ---------------------------------------------------------------------------
__global__ void __launch_bounds__(128) attn_kernel(
    const bf16* __restrict__ Qb, const bf16* __restrict__ Kb,
    const bf16* __restrict__ Vt, bf16* __restrict__ Ob) {
  __shared__ bf16 pbuf[4][16 * 64];            // 2 KB per wave
  const int lane  = threadIdx.x & 31;
  const int wave  = threadIdx.x >> 5;
  const int lhalf = lane & 15;
  const int koff  = (lane < 16) ? 0 : 8;
  const int rofs  = (lane < 16) ? 0 : 8;
  const int bh = blockIdx.y;                   // b*16 + h
  const int q0 = blockIdx.x * 64 + wave * 16;

  const bf16* Qp = Qb + (size_t)bh * Sq * HDd;
  const bf16* Kp = Kb + (size_t)bh * Sq * HDd;
  const bf16* Vp = Vt + (size_t)bh * HDd * Sq;

  // Q tile A-operands (two hd chunks of 32), reused for all key rounds
  const bf16* qrow = Qp + (size_t)(q0 + lhalf) * HDd;
  v16bf aq0 = ldab(qrow + koff);
  v16bf aq1 = ldab(qrow + 32 + koff);

  float mrun[8], lrun[8];
#pragma unroll
  for (int i = 0; i < 8; i++) { mrun[i] = -3.0e38f; lrun[i] = 0.f; }
  v8f o[4];
#pragma unroll
  for (int j = 0; j < 4; j++)
#pragma unroll
    for (int e = 0; e < 8; e++) o[j][e] = 0.f;

  bf16* pb = pbuf[wave];

  for (int kt = 0; kt < Sq; kt += 64) {
    // ---- scores: four 16-key chunks, contraction over hd=64 ----
    v8f sc[4];
#pragma unroll
    for (int c = 0; c < 4; c++) {
      const bf16* krow = Kp + (size_t)(kt + c * 16 + lhalf) * HDd;
      v16bf b0 = ldab(krow + koff);
      v16bf b1 = ldab(krow + 32 + koff);
      v8f zz;
#pragma unroll
      for (int e = 0; e < 8; e++) zz[e] = 0.f;
      zz = wmma_bf16(aq0, b0, zz);
      sc[c] = wmma_bf16(aq1, b1, zz);
    }

    // ---- online softmax: row stats via 16-lane-half shfl reductions ----
    float alpha[8];
#pragma unroll
    for (int i = 0; i < 8; i++) {
      float m = fmaxf(fmaxf(sc[0][i], sc[1][i]), fmaxf(sc[2][i], sc[3][i]));
#pragma unroll
      for (int msk = 8; msk; msk >>= 1) m = fmaxf(m, __shfl_xor(m, msk, 32));
      const float mnew = fmaxf(mrun[i], m);
      alpha[i] = __expf(mrun[i] - mnew);
      mrun[i] = mnew;
      float rs = 0.f;
#pragma unroll
      for (int c = 0; c < 4; c++) {
        const float p = __expf(sc[c][i] - mnew);
        sc[c][i] = p;
        rs += p;
      }
#pragma unroll
      for (int msk = 8; msk; msk >>= 1) rs += __shfl_xor(rs, msk, 32);
      lrun[i] = lrun[i] * alpha[i] + rs;
    }
#pragma unroll
    for (int j = 0; j < 4; j++)
#pragma unroll
      for (int e = 0; e < 8; e++) o[j][e] *= alpha[e];

    // ---- P: D-layout -> LDS row-major 16x64 -> reload as A-operands ----
#pragma unroll
    for (int c = 0; c < 4; c++)
#pragma unroll
      for (int i = 0; i < 8; i++)
        pb[(i + rofs) * 64 + c * 16 + lhalf] = f2bf(sc[c][i]);
    __syncthreads();
    v16bf ap0 = ldab(pb + lhalf * 64 + koff);        // keys kt..kt+31
    v16bf ap1 = ldab(pb + lhalf * 64 + 32 + koff);   // keys kt+32..kt+63
    __syncthreads();

    if (kt + 64 < Sq)   // warm next K tile (global_prefetch_b8)
      __builtin_prefetch(Kp + (size_t)(kt + 64 + lhalf) * HDd, 0, 1);

    // ---- PV: V transposed so keys are contiguous per hd column ----
#pragma unroll
    for (int j = 0; j < 4; j++) {
      const bf16* vrow = Vp + (size_t)(j * 16 + lhalf) * Sq + kt;
      v16bf b0 = ldab(vrow + koff);
      v16bf b1 = ldab(vrow + 32 + koff);
      o[j] = wmma_bf16(ap0, b0, o[j]);
      o[j] = wmma_bf16(ap1, b1, o[j]);
    }
  }

  // ---- normalize + write attended [b][s][h*64+hd] as bf16 ----
  const int h = bh & 15, bidx = bh >> 4;
#pragma unroll
  for (int j = 0; j < 4; j++) {
    const int hd = j * 16 + lhalf;
#pragma unroll
    for (int e = 0; e < 8; e++) {
      const int s = q0 + e + rofs;
      const float v = o[j][e] / lrun[e];
      Ob[(size_t)(bidx * Sq + s) * Dm + h * 64 + hd] = f2bf(v);
    }
  }
}

// ---------------------------------------------------------------------------
// Output projection: attended(bf16) x WoT + bo -> f32 d_out.
// Same 32x64-per-wave tiling as the QKV GEMM.
// ---------------------------------------------------------------------------
__global__ void __launch_bounds__(128) out_gemm_kernel(
    const bf16* __restrict__ X, const bf16* __restrict__ Wt,
    const float* __restrict__ bias, float* __restrict__ out) {
  const int lane  = threadIdx.x & 31;
  const int wave  = threadIdx.x >> 5;
  const int lhalf = lane & 15;
  const int koff  = (lane < 16) ? 0 : 8;
  const int m0 = blockIdx.x * 128 + wave * 32;
  const int n0 = blockIdx.y * 64;

  v8f acc[2][4];
#pragma unroll
  for (int r = 0; r < 2; r++)
#pragma unroll
    for (int j = 0; j < 4; j++)
#pragma unroll
      for (int e = 0; e < 8; e++) acc[r][j][e] = 0.f;

  const bf16* arow0 = X + (size_t)(m0 + lhalf) * Dm;
  const bf16* arow1 = X + (size_t)(m0 + 16 + lhalf) * Dm;
  for (int k0 = 0; k0 < Dm; k0 += 32) {
    v16bf a0 = ldab(arow0 + k0 + koff);
    v16bf a1 = ldab(arow1 + k0 + koff);
#pragma unroll
    for (int j = 0; j < 4; j++) {
      const bf16* brow = Wt + (size_t)(n0 + j * 16 + lhalf) * Dm;
      v16bf b = ldab(brow + k0 + koff);
      acc[0][j] = wmma_bf16(a0, b, acc[0][j]);
      acc[1][j] = wmma_bf16(a1, b, acc[1][j]);
    }
  }

#pragma unroll
  for (int r = 0; r < 2; r++) {
    const int rbase = m0 + r * 16 + ((lane < 16) ? 0 : 8);
#pragma unroll
    for (int j = 0; j < 4; j++) {
      const int col = n0 + j * 16 + lhalf;
      const float bb = bias[col];
#pragma unroll
      for (int e = 0; e < 8; e++)
        out[(size_t)(rbase + e) * Dm + col] = acc[r][j][e] + bb;
    }
  }
}

// ---------------------------------------------------------------------------
extern "C" void kernel_launch(void* const* d_in, const int* in_sizes, int n_in,
                              void* d_out, int out_size, void* d_ws, size_t ws_size,
                              hipStream_t stream) {
  (void)in_sizes; (void)n_in; (void)out_size; (void)ws_size;
  const float* x  = (const float*)d_in[0];
  const float* Wq = (const float*)d_in[1];
  const float* bq = (const float*)d_in[2];
  const float* Wk = (const float*)d_in[3];
  const float* bk = (const float*)d_in[4];
  const float* Wv = (const float*)d_in[5];
  const float* bv = (const float*)d_in[6];
  const float* Wo = (const float*)d_in[7];
  const float* bo = (const float*)d_in[8];
  float* out = (float*)d_out;

  char* ws = (char*)d_ws;
  const size_t MiB = 1ull << 20;
  bf16* xb  = (bf16*)(ws + 0);          // 8 MiB  x in bf16
  bf16* WqT = (bf16*)(ws + 8  * MiB);   // 2 MiB each, N-major
  bf16* WkT = (bf16*)(ws + 10 * MiB);
  bf16* WvT = (bf16*)(ws + 12 * MiB);
  bf16* WoT = (bf16*)(ws + 14 * MiB);
  bf16* Qb  = (bf16*)(ws + 16 * MiB);   // 8 MiB [bh][s][hd] (pre-scaled)
  bf16* Kb  = (bf16*)(ws + 24 * MiB);   // 8 MiB [bh][s][hd]
  bf16* Vt  = (bf16*)(ws + 32 * MiB);   // 8 MiB [bh][hd][s]
  bf16* Ab  = (bf16*)(ws + 40 * MiB);   // 8 MiB attended bf16

  cvt_kernel<<<2048, 256, 0, stream>>>(x, xb, Mtot * Dm);
  dim3 tb(32, 8), tg(Dm / 32, Dm / 32);
  transpose_cvt_kernel<<<tg, tb, 0, stream>>>(Wq, WqT);
  transpose_cvt_kernel<<<tg, tb, 0, stream>>>(Wk, WkT);
  transpose_cvt_kernel<<<tg, tb, 0, stream>>>(Wv, WvT);
  transpose_cvt_kernel<<<tg, tb, 0, stream>>>(Wo, WoT);

  // Fused QKV: grid.z selects weight; 128x64 workgroup tile (4 waves).
  qkv_gemm_kernel<<<dim3(Mtot / 128, Dm / 64, 3), 128, 0, stream>>>(
      xb, WqT, WkT, WvT, bq, bk, bv, Qb, Kb, Vt);

  attn_kernel<<<dim3(Sq / 64, Bq * Hh), 128, 0, stream>>>(Qb, Kb, Vt, Ab);

  out_gemm_kernel<<<dim3(Mtot / 128, Dm / 64), 128, 0, stream>>>(Ab, WoT, bo, out);
}
